// LazyMLPBlock_81381040325097
// MI455X (gfx1250) — compile-verified, matrix operand order checked by reading, not run on hardware
//
#include <hip/hip_runtime.h>
#include <hip/hip_bf16.h>
#include <stdint.h>

#define T_TOK 256
#define H_DIM 2048
#define I_DIM 2048
#define E_NUM 16
#define TOPK  4
#define TWO_I 4096
#define LIMIT 7.0f
#define ALPHA 1.702f
#define EPS   1e-5f

#define KSTEP 64              // K per double-buffer stage (2 WMMAs deep)
#define BROWS 128             // output columns per block (8 waves x 16)
#define RTMAX 4               // row-tiles (16 rows each) per block: 64-row super-tile
#define BPADF 68              // 64 floats + 4 pad  -> 272B rows (16B aligned)
#define APADH 72              // 64 bf16  + 8 pad   -> 144B rows (16B aligned)

typedef __attribute__((ext_vector_type(16))) __bf16 v16bf;
typedef __attribute__((ext_vector_type(8)))  float  v8f;

__device__ __forceinline__ unsigned lds_off(const void* p) {
  // generic pointer to LDS: low 32 bits are the wave-relative LDS byte offset
  return (unsigned)(uintptr_t)p;
}
__device__ __forceinline__ void async_b128(unsigned lds, const void* g) {
  asm volatile("global_load_async_to_lds_b128 %0, %1, off"
               :: "v"(lds), "v"((unsigned long long)(uintptr_t)g) : "memory");
}
__device__ __forceinline__ void wait_async0() {
#if __has_builtin(__builtin_amdgcn_s_wait_asynccnt)
  __builtin_amdgcn_s_wait_asynccnt(0);
#else
  asm volatile("s_wait_asynccnt 0x0" ::: "memory");
#endif
}

// ---------------------------------------------------------------------------
// Kernel 1: fused RMSNorm + gate logits + top-4 + softmax.  One block / token.
// ---------------------------------------------------------------------------
__global__ __launch_bounds__(256)
void rmsnorm_gate_topk(const float* __restrict__ x,
                       const float* __restrict__ nscale,
                       const float* __restrict__ gate_w,
                       __bf16* __restrict__ t_bf,
                       float* __restrict__ ew,
                       int* __restrict__ idx) {
  __shared__ float red[256];
  __shared__ float logits[E_NUM];
  const int tid = threadIdx.x, tok = blockIdx.x;
  const float* xr = x + (size_t)tok * H_DIM;

  float v[8]; float ssq = 0.f;
#pragma unroll
  for (int s = 0; s < 8; ++s) { v[s] = xr[tid + 256 * s]; ssq += v[s] * v[s]; }
  red[tid] = ssq; __syncthreads();
  for (int off = 128; off > 0; off >>= 1) {
    if (tid < off) red[tid] += red[tid + off];
    __syncthreads();
  }
  const float rinv = rsqrtf(red[0] / (float)H_DIM + EPS);
  if (tid < E_NUM) logits[tid] = 0.f;
  __syncthreads();

  float tf[8];
#pragma unroll
  for (int s = 0; s < 8; ++s) {
    const int j = tid + 256 * s;
    tf[s] = v[s] * rinv * nscale[j];
    t_bf[(size_t)tok * H_DIM + j] = (__bf16)tf[s];
  }
  for (int e = 0; e < E_NUM; ++e) {
    const float* gw = gate_w + (size_t)e * H_DIM;
    float p = 0.f;
#pragma unroll
    for (int s = 0; s < 8; ++s) p += tf[s] * gw[tid + 256 * s];
    atomicAdd(&logits[e], p);
  }
  __syncthreads();

  if (tid == 0) {
    float lg[E_NUM];
    for (int e = 0; e < E_NUM; ++e) lg[e] = logits[e];
    float vals[TOPK]; int ids[TOPK];
    for (int k = 0; k < TOPK; ++k) {
      float best = -1e30f; int bi = 0;
      for (int e = 0; e < E_NUM; ++e) if (lg[e] > best) { best = lg[e]; bi = e; }
      vals[k] = best; ids[k] = bi; lg[bi] = -1e30f;
    }
    const float mx = vals[0];
    float w[TOPK], sum = 0.f;
    for (int k = 0; k < TOPK; ++k) { w[k] = __expf(vals[k] - mx); sum += w[k]; }
    for (int k = 0; k < TOPK; ++k) {
      ew[tok * TOPK + k] = w[k] / sum;
      idx[tok * TOPK + k] = ids[k];
    }
  }
}

// ---------------------------------------------------------------------------
// Kernel 2: deterministic per-expert compaction of the 1024 (token,slot) pairs
// ---------------------------------------------------------------------------
__global__ void route_compact(const int* __restrict__ idx,
                              int* __restrict__ list,
                              int* __restrict__ counts,
                              int* __restrict__ offsets) {
  const int e = threadIdx.x;
  if (e < E_NUM) {
    int cnt = 0;
    for (int p = 0; p < T_TOK * TOPK; ++p)
      if (idx[p] == e) list[e * T_TOK + cnt++] = p;   // pair id = token*4 + slot
    counts[e] = cnt;
  }
  __syncthreads();
  if (e == 0) {
    int run = 0;
    for (int i = 0; i < E_NUM; ++i) { offsets[i] = run; run += counts[i]; }
  }
}

// ---------------------------------------------------------------------------
// Shared WMMA helpers
// ---------------------------------------------------------------------------
__device__ __forceinline__ v16bf make_bfrag_f32(const float* row, int base) {
  const float4 f0 = *(const float4*)(row + base);
  const float4 f1 = *(const float4*)(row + base + 4);
  const float4 f2 = *(const float4*)(row + base + 16);
  const float4 f3 = *(const float4*)(row + base + 20);
  union { v16bf v; __bf16 e[16]; } B;
  B.e[0]  = (__bf16)f0.x; B.e[1]  = (__bf16)f0.y;
  B.e[2]  = (__bf16)f0.z; B.e[3]  = (__bf16)f0.w;
  B.e[4]  = (__bf16)f1.x; B.e[5]  = (__bf16)f1.y;
  B.e[6]  = (__bf16)f1.z; B.e[7]  = (__bf16)f1.w;
  B.e[8]  = (__bf16)f2.x; B.e[9]  = (__bf16)f2.y;
  B.e[10] = (__bf16)f2.z; B.e[11] = (__bf16)f2.w;
  B.e[12] = (__bf16)f3.x; B.e[13] = (__bf16)f3.y;
  B.e[14] = (__bf16)f3.z; B.e[15] = (__bf16)f3.w;
  return B.v;
}
__device__ __forceinline__ v16bf make_afrag_bf16(const __bf16* row, int base) {
  union { v16bf v; uint4 q[2]; } A;
  A.q[0] = *(const uint4*)(row + base);
  A.q[1] = *(const uint4*)(row + base + 16);
  return A.v;
}

// ---------------------------------------------------------------------------
// Kernel 3: grouped GEMM1  h = t @ w1[e]^T + b1 ; fused SwiGLU -> a (bf16)
// grid = (2I/128, 4, E); block = 256 (8 waves).  Each block owns a 64-row
// super-tile; 4 accumulators per wave reuse each converted B fragment 4x.
// WMMA loop is branch-free (padding rows are clamped duplicates, discarded at
// store time) so the scheduler can pipeline ds_loads under the WMMAs.
// ---------------------------------------------------------------------------
__global__ __launch_bounds__(256)
void gemm1_swiglu(const __bf16* __restrict__ t_bf,
                  const float* __restrict__ w1,
                  const float* __restrict__ b1,
                  const int* __restrict__ counts,
                  const int* __restrict__ offsets,
                  const int* __restrict__ list,
                  __bf16* __restrict__ a_bf) {
  const int e = blockIdx.z;
  const int nrows = counts[e];
  const int row0 = blockIdx.y * (RTMAX * 16);
  if (row0 >= nrows) return;
  const int rt_n = min(RTMAX, (nrows - row0 + 15) >> 4);  // for store epilogue
  const int col0 = blockIdx.x * BROWS;

  __shared__ __bf16 aT[2][RTMAX * 16][APADH];   // 64 rows x 64 bf16 (+pad)
  __shared__ float  bT[2][BROWS][BPADF];        // 128 rows x 64 fp32 (+pad)
  __shared__ int toks[RTMAX * 16];

  const int tid = threadIdx.x;
  if (tid < 64) {
    int gr = row0 + tid;
    if (gr >= nrows) gr = nrows - 1;
    toks[tid] = list[e * T_TOK + gr] >> 2;      // token id (clamped duplicate)
  }
  __syncthreads();

  const int lane = tid & 31, wave = tid >> 5;
  const int kg = lane >> 4, mn = lane & 15, k0 = kg * 8;
  const int cl = wave * 16 + mn;                // local output column
  v8f acc[RTMAX] = {};

  auto stage = [&](int p, int kb) {
#pragma unroll
    for (int i = 0; i < 2; ++i) {               // A: 64 rows x 64 bf16
      const int flat = tid + 256 * i;
      const int r = flat >> 3, s8 = flat & 7;
      async_b128(lds_off(&aT[p][r][s8 * 8]),
                 t_bf + (size_t)toks[r] * H_DIM + kb + s8 * 8);
    }
#pragma unroll
    for (int i = 0; i < 8; ++i) {               // B: 128 rows x 64 fp32
      const int flat = tid + 256 * i;
      const int r = flat >> 4, s4 = flat & 15;
      async_b128(lds_off(&bT[p][r][s4 * 4]),
                 w1 + ((size_t)e * TWO_I + col0 + r) * H_DIM + kb + s4 * 4);
    }
  };
  auto compute = [&](int p) {
#pragma unroll
    for (int half = 0; half < 2; ++half) {
      const v16bf B = make_bfrag_f32(&bT[p][cl][0], half * 32 + k0);
#pragma unroll
      for (int rt = 0; rt < RTMAX; ++rt) {      // branch-free
        const v16bf A = make_afrag_bf16(&aT[p][rt * 16 + mn][0], half * 32 + k0);
        acc[rt] = __builtin_amdgcn_wmma_f32_16x16x32_bf16(
            false, A, false, B, (short)0, acc[rt], false, false);
      }
    }
  };

  stage(0, 0);
  for (int kt = 0; kt < H_DIM / KSTEP; ++kt) {
    wait_async0();                              // my asyncs for buf kt&1 done
    __syncthreads();                            // => everyone's are done
    if (kt + 1 < H_DIM / KSTEP) stage((kt + 1) & 1, (kt + 1) * KSTEP);
    compute(kt & 1);
  }

  // Epilogue: + b1, interleaved SwiGLU (even ch = glu, odd = lin), store a bf16
  const int cg = col0 + cl;
  const float b1v = b1[(size_t)e * TWO_I + cg];
  const int aoff = offsets[e];
  const bool isglu = ((cg & 1) == 0);
#pragma unroll
  for (int rt = 0; rt < RTMAX; ++rt) {
    if (rt < rt_n) {
#pragma unroll
      for (int v = 0; v < 8; ++v) {
        const float h = acc[rt][v] + b1v;
        const float partner = __shfl_xor(h, 1, 32);  // pairs lanes (2j, 2j+1)
        const int gr = row0 + rt * 16 + v + 8 * kg;
        if (isglu && gr < nrows) {
          const float glu = fminf(h, LIMIT);
          const float lin = fminf(fmaxf(partner, -LIMIT), LIMIT);
          const float av = glu * (1.f / (1.f + __expf(-ALPHA * glu))) * (lin + 1.f);
          a_bf[(size_t)(aoff + gr) * I_DIM + (cg >> 1)] = (__bf16)av;
        }
      }
    }
  }
}

// ---------------------------------------------------------------------------
// Kernel 4: grouped GEMM2  o = a @ w2[e]^T + b2, scaled by router weight,
// scattered to contrib[(token*4+slot)*H + c].  grid = (H/128, 4, E)
// ---------------------------------------------------------------------------
__global__ __launch_bounds__(256)
void gemm2_scatter(const __bf16* __restrict__ a_bf,
                   const float* __restrict__ w2,
                   const float* __restrict__ b2,
                   const int* __restrict__ counts,
                   const int* __restrict__ offsets,
                   const int* __restrict__ list,
                   const float* __restrict__ ew,
                   float* __restrict__ contrib) {
  const int e = blockIdx.z;
  const int nrows = counts[e];
  const int row0 = blockIdx.y * (RTMAX * 16);
  if (row0 >= nrows) return;
  const int rt_n = min(RTMAX, (nrows - row0 + 15) >> 4);
  const int col0 = blockIdx.x * BROWS;
  const int aoff = offsets[e];

  __shared__ __bf16 aT[2][RTMAX * 16][APADH];
  __shared__ float  bT[2][BROWS][BPADF];
  __shared__ int   arow[RTMAX * 16];            // clamped activation row index
  __shared__ int   pids[RTMAX * 16];
  __shared__ float wts[RTMAX * 16];

  const int tid = threadIdx.x;
  if (tid < 64) {
    int gr = row0 + tid;
    if (gr >= nrows) gr = nrows - 1;
    arow[tid] = aoff + gr;
    const int pid = list[e * T_TOK + gr];
    pids[tid] = pid;
    wts[tid] = ew[pid];
  }
  __syncthreads();

  const int lane = tid & 31, wave = tid >> 5;
  const int kg = lane >> 4, mn = lane & 15, k0 = kg * 8;
  const int cl = wave * 16 + mn;
  v8f acc[RTMAX] = {};

  auto stage = [&](int p, int kb) {
#pragma unroll
    for (int i = 0; i < 2; ++i) {               // A: activations (already bf16)
      const int flat = tid + 256 * i;
      const int r = flat >> 3, s8 = flat & 7;
      async_b128(lds_off(&aT[p][r][s8 * 8]),
                 a_bf + (size_t)arow[r] * I_DIM + kb + s8 * 8);
    }
#pragma unroll
    for (int i = 0; i < 8; ++i) {               // B: w2 fp32
      const int flat = tid + 256 * i;
      const int r = flat >> 4, s4 = flat & 15;
      async_b128(lds_off(&bT[p][r][s4 * 4]),
                 w2 + ((size_t)e * H_DIM + col0 + r) * I_DIM + kb + s4 * 4);
    }
  };
  auto compute = [&](int p) {
#pragma unroll
    for (int half = 0; half < 2; ++half) {
      const v16bf B = make_bfrag_f32(&bT[p][cl][0], half * 32 + k0);
#pragma unroll
      for (int rt = 0; rt < RTMAX; ++rt) {      // branch-free
        const v16bf A = make_afrag_bf16(&aT[p][rt * 16 + mn][0], half * 32 + k0);
        acc[rt] = __builtin_amdgcn_wmma_f32_16x16x32_bf16(
            false, A, false, B, (short)0, acc[rt], false, false);
      }
    }
  };

  stage(0, 0);
  for (int kt = 0; kt < I_DIM / KSTEP; ++kt) {
    wait_async0();
    __syncthreads();
    if (kt + 1 < I_DIM / KSTEP) stage((kt + 1) & 1, (kt + 1) * KSTEP);
    compute(kt & 1);
  }

  const int cg = col0 + cl;
  const float b2v = b2[(size_t)e * H_DIM + cg];
#pragma unroll
  for (int rt = 0; rt < RTMAX; ++rt) {
    if (rt < rt_n) {
#pragma unroll
      for (int v = 0; v < 8; ++v) {
        const int m = rt * 16 + v + 8 * kg;
        const int gr = row0 + m;
        if (gr < nrows) {
          contrib[(size_t)pids[m] * H_DIM + cg] = (acc[rt][v] + b2v) * wts[m];
        }
      }
    }
  }
}

// ---------------------------------------------------------------------------
// Kernel 5: out = x + sum_k contrib[token,k]
// ---------------------------------------------------------------------------
__global__ __launch_bounds__(256)
void finalize_sum(const float* __restrict__ x,
                  const float* __restrict__ contrib,
                  float* __restrict__ out) {
  const int i = blockIdx.x * 256 + threadIdx.x;
  const int t = i / H_DIM, c = i % H_DIM;
  float s = x[i];
#pragma unroll
  for (int k = 0; k < TOPK; ++k) s += contrib[((size_t)t * TOPK + k) * H_DIM + c];
  out[i] = s;
}

// ---------------------------------------------------------------------------
extern "C" void kernel_launch(void* const* d_in, const int* in_sizes, int n_in,
                              void* d_out, int out_size, void* d_ws, size_t ws_size,
                              hipStream_t stream) {
  const float* x      = (const float*)d_in[0];
  const float* nscale = (const float*)d_in[1];
  const float* gate_w = (const float*)d_in[2];
  const float* w1     = (const float*)d_in[3];
  const float* b1     = (const float*)d_in[4];
  const float* w2     = (const float*)d_in[5];
  const float* b2     = (const float*)d_in[6];
  float* out = (float*)d_out;

  char* ws = (char*)d_ws;
  auto alloc = [&](size_t sz) {
    char* p = ws;
    ws += (sz + 255) & ~(size_t)255;
    return p;
  };
  __bf16*  t_bf    = (__bf16*)alloc((size_t)T_TOK * H_DIM * 2);
  float*   ewp     = (float*)alloc((size_t)T_TOK * TOPK * 4);
  int*     idxp    = (int*)alloc((size_t)T_TOK * TOPK * 4);
  int*     counts  = (int*)alloc(E_NUM * 4);
  int*     offs    = (int*)alloc(E_NUM * 4);
  int*     list    = (int*)alloc((size_t)E_NUM * T_TOK * 4);
  __bf16*  a_bf    = (__bf16*)alloc((size_t)T_TOK * TOPK * I_DIM * 2);
  float*   contrib = (float*)alloc((size_t)T_TOK * TOPK * H_DIM * 4);

  rmsnorm_gate_topk<<<T_TOK, 256, 0, stream>>>(x, nscale, gate_w, t_bf, ewp, idxp);
  route_compact<<<1, 32, 0, stream>>>(idxp, list, counts, offs);
  gemm1_swiglu<<<dim3(TWO_I / BROWS, RTMAX, E_NUM), 256, 0, stream>>>(
      t_bf, w1, b1, counts, offs, list, a_bf);
  gemm2_scatter<<<dim3(H_DIM / BROWS, RTMAX, E_NUM), 256, 0, stream>>>(
      a_bf, w2, b2, counts, offs, list, ewp, contrib);
  finalize_sum<<<(T_TOK * H_DIM) / 256, 256, 0, stream>>>(x, contrib, out);
}
